// Kernel_61229053771823
// MI455X (gfx1250) — compile-verified, matrix operand order checked by reading, not run on hardware
//
#include <hip/hip_runtime.h>
#include <math.h>

// ---------------------------------------------------------------------------
// RBF kernel out[b,n] = exp(-0.5*|x_b - w_n|^2)
//   = exp( cross[b,n] - 0.5*x2[b] - 0.5*w2[n] ),  cross = X * W^T
// cross via V_WMMA_F32_16X16X32_BF16 with bf16x3 split (hi/lo) for
// fp32-equivalent accuracy on the fast 16-bit matrix pipe (gfx1250, wave32).
// Fast path: one-pass fp32->bf16(hi/lo) split into d_ws, then a GEMM whose
// global->LDS traffic uses CDNA5 async copies (global_load_async_to_lds_b128
// + s_wait_asynccnt). Fallback path converts in-kernel (needs only 48KB ws).
// ---------------------------------------------------------------------------

typedef __attribute__((ext_vector_type(16))) __bf16        v16bf;
typedef __attribute__((ext_vector_type(8)))  float         v8f;
typedef __attribute__((ext_vector_type(8)))  unsigned int  v8u;

#define MDIM 512      // K dimension (feature dim)
#define NDIM 8192     // columns of output (rows of W)
#define BDIM 4096     // rows of output (rows of X)
#define TILE 128      // block macro-tile (rows and cols)
#define KC   32       // K-chunk = one bf16 WMMA K step
#define KDW  16       // dwords (bf16 pairs) per row per chunk
#define LDSW 17       // fallback kernel LDS row stride (dwords)
#define LDSWA 20      // async kernel LDS row stride: 16B-aligned + conflict-free
#define REGSZ (TILE * LDSWA)   // 2560 dwords per LDS region

__device__ __forceinline__ unsigned short bf16_rne(float f) {
    unsigned int u = __builtin_bit_cast(unsigned int, f);
    u += 0x7FFFu + ((u >> 16) & 1u);          // round-to-nearest-even
    return (unsigned short)(u >> 16);
}

// Split two consecutive f32 into packed bf16-hi dword and bf16-lo dword.
__device__ __forceinline__ void split2(float a, float b,
                                       unsigned int& hi, unsigned int& lo) {
    unsigned short ha = bf16_rne(a), hb = bf16_rne(b);
    float fa = __builtin_bit_cast(float, (unsigned int)ha << 16);
    float fb = __builtin_bit_cast(float, (unsigned int)hb << 16);
    unsigned short la = bf16_rne(a - fa), lb = bf16_rne(b - fb);
    hi = (unsigned int)ha | ((unsigned int)hb << 16);
    lo = (unsigned int)la | ((unsigned int)lb << 16);
}

// CDNA5 async global->LDS copy, 16B per lane (GVS mode).
// vdst = LDS byte address (low 32 bits of generic LDS pointer),
// vaddr = 32-bit byte offset, saddr = 64-bit base.
__device__ __forceinline__ void async_copy_b128(const unsigned int* lds_ptr,
                                                unsigned int goff_byte,
                                                const void* base) {
    unsigned int lds_byte = (unsigned int)(unsigned long long)lds_ptr;
    asm volatile("global_load_async_to_lds_b128 %0, %1, %2"
                 :: "v"(lds_byte), "v"(goff_byte), "s"(base)
                 : "memory");
}
__device__ __forceinline__ void wait_async0() {
    asm volatile("s_wait_asynccnt 0" ::: "memory");
}

// ================== FAST PATH: one-pass split + row norms ==================
__global__ __launch_bounds__(256) void split_norm_kernel(
        const float* __restrict__ X, const float* __restrict__ W,
        float* __restrict__ x2, float* __restrict__ w2,
        unsigned int* __restrict__ Xhi, unsigned int* __restrict__ Xlo,
        unsigned int* __restrict__ Whi, unsigned int* __restrict__ Wlo) {
    const int row  = blockIdx.x * 8 + (threadIdx.x >> 5);
    const int lane = threadIdx.x & 31;
    const float* src;
    float* ndst;
    unsigned int *hdst, *ldst;
    if (row < BDIM) {
        src = X + (size_t)row * MDIM; ndst = x2 + row;
        hdst = Xhi + (size_t)row * (MDIM / 2);
        ldst = Xlo + (size_t)row * (MDIM / 2);
    } else {
        int r = row - BDIM;
        src = W + (size_t)r * MDIM; ndst = w2 + r;
        hdst = Whi + (size_t)r * (MDIM / 2);
        ldst = Wlo + (size_t)r * (MDIM / 2);
    }
    float s = 0.f;
    #pragma unroll
    for (int i = 0; i < MDIM / 64; ++i) {      // 8 dword-pairs per lane
        int j = lane + i * 32;                 // pair index 0..255
        float2 v = *(const float2*)&src[2 * j];
        unsigned int h, l;
        split2(v.x, v.y, h, l);
        hdst[j] = h; ldst[j] = l;
        s += v.x * v.x + v.y * v.y;
    }
    #pragma unroll
    for (int o = 16; o; o >>= 1) s += __shfl_xor(s, o, 32);
    if (lane == 0) *ndst = s;
}

// ============== FAST PATH: WMMA GEMM fed by async LDS copies ===============
__global__ __launch_bounds__(256) void rbf_wmma_async_kernel(
        const unsigned int* __restrict__ Xhi, const unsigned int* __restrict__ Xlo,
        const unsigned int* __restrict__ Whi, const unsigned int* __restrict__ Wlo,
        const float* __restrict__ x2, const float* __restrict__ w2,
        float* __restrict__ out) {
    // [Xhi | Xlo | Whi | Wlo] + 256-dword padding that receives a visible
    // per-thread store: this keeps the optimizer from proving smem is never
    // written (the real writers are opaque async-copy asm) and folding the
    // fragment ds_loads to undef.
    __shared__ unsigned int smem[4 * REGSZ + 256];

    const int t     = threadIdx.x;
    smem[4 * REGSZ + t] = (unsigned int)t;     // visible store; never read

    const int lane  = t & 31;
    const int wave  = t >> 5;
    const int l16   = lane & 15;
    const int half  = lane >> 4;
    const int waveM = wave & 3;                // 4 wave-rows * 32 rows
    const int waveN = wave >> 2;               // 2 wave-cols * 64 cols
    const int rowBase = blockIdx.y * TILE;
    const int colBase = blockIdx.x * TILE;

    const unsigned int* gbase[4] = {Xhi, Xlo, Whi, Wlo};
    const int           rbase[4] = {rowBase, rowBase, colBase, colBase};

    v8f acc[2][4];
    const v8f vzero = {0.f, 0.f, 0.f, 0.f, 0.f, 0.f, 0.f, 0.f};
    #pragma unroll
    for (int mi = 0; mi < 2; ++mi)
        #pragma unroll
        for (int ni = 0; ni < 4; ++ni) acc[mi][ni] = vzero;

    auto frag = [&](int region, int row) -> v16bf {
        const unsigned int* rp = smem + region * REGSZ + row * LDSWA;
        v8u u;
        #pragma unroll
        for (int v = 0; v < 8; ++v)
            u[v] = rp[(v & 3) + half * 4 + (v >> 2) * 8];
        return __builtin_bit_cast(v16bf, u);
    };

    const int NCH = MDIM / KC;   // 16 K-chunks
    for (int kc = 0; kc < NCH; ++kc) {
        // ---- async-copy this chunk's 4 regions (128 rows x 16 dwords each)
        // 2048 b128 transfers total -> 8 per thread
        #pragma unroll
        for (int i = 0; i < 8; ++i) {
            int region = i >> 1;
            int q   = (i & 1) * 256 + t;       // 0..511 slot within region
            int row = q >> 2;
            int grp = q & 3;                   // 4 dwords per b128
            const unsigned int* lds_ptr = smem + region * REGSZ + row * LDSWA + grp * 4;
            unsigned int goff = (unsigned int)((rbase[region] + row) * (MDIM / 2)
                                               + kc * KDW + grp * 4) * 4u;
            async_copy_b128(lds_ptr, goff, gbase[region]);
        }
        wait_async0();        // own transfers landed in LDS
        __syncthreads();      // all waves' transfers landed

        v16bf Ah[2], Al[2], Bh[4], Bl[4];
        #pragma unroll
        for (int mi = 0; mi < 2; ++mi) {
            int r = waveM * 32 + mi * 16 + l16;
            Ah[mi] = frag(0, r);
            Al[mi] = frag(1, r);
        }
        #pragma unroll
        for (int ni = 0; ni < 4; ++ni) {
            int r = waveN * 64 + ni * 16 + l16;
            Bh[ni] = frag(2, r);
            Bl[ni] = frag(3, r);
        }
        #pragma unroll
        for (int mi = 0; mi < 2; ++mi) {
            #pragma unroll
            for (int ni = 0; ni < 4; ++ni) {
                acc[mi][ni] = __builtin_amdgcn_wmma_f32_16x16x32_bf16(
                    false, Ah[mi], false, Bh[ni], (short)0, acc[mi][ni], false, false);
                acc[mi][ni] = __builtin_amdgcn_wmma_f32_16x16x32_bf16(
                    false, Ah[mi], false, Bl[ni], (short)0, acc[mi][ni], false, false);
                acc[mi][ni] = __builtin_amdgcn_wmma_f32_16x16x32_bf16(
                    false, Al[mi], false, Bh[ni], (short)0, acc[mi][ni], false, false);
            }
        }
        __syncthreads();      // all reads done before next chunk overwrites
    }

    // Epilogue: out = exp(cross - 0.5*x2 - 0.5*w2); C layout: M = r + 8*half, N = l16
    #pragma unroll
    for (int mi = 0; mi < 2; ++mi) {
        #pragma unroll
        for (int ni = 0; ni < 4; ++ni) {
            const int n = colBase + waveN * 64 + ni * 16 + l16;
            const float w2v = 0.5f * w2[n];
            #pragma unroll
            for (int r = 0; r < 8; ++r) {
                const int m = rowBase + waveM * 32 + mi * 16 + r + 8 * half;
                const float d = acc[mi][ni][r] - 0.5f * x2[m] - w2v;
                out[(size_t)m * NDIM + n] = expf(d);
            }
        }
    }
}

// ================= FALLBACK PATH (small ws): round-0 kernels ===============
__global__ __launch_bounds__(256) void rownorm_kernel(
        const float* __restrict__ X, const float* __restrict__ W,
        float* __restrict__ x2, float* __restrict__ w2) {
    const int row  = blockIdx.x * 8 + (threadIdx.x >> 5);
    const int lane = threadIdx.x & 31;
    const float* src;
    float* dst;
    if (row < BDIM) { src = X + (size_t)row * MDIM;          dst = x2 + row; }
    else            { src = W + (size_t)(row - BDIM) * MDIM; dst = w2 + (row - BDIM); }
    float s = 0.f;
    #pragma unroll
    for (int k = lane; k < MDIM; k += 32) { float v = src[k]; s += v * v; }
    #pragma unroll
    for (int o = 16; o; o >>= 1) s += __shfl_xor(s, o, 32);
    if (lane == 0) *dst = s;
}

__global__ __launch_bounds__(256) void rbf_wmma_kernel(
        const float* __restrict__ X, const float* __restrict__ W,
        const float* __restrict__ x2, const float* __restrict__ w2,
        float* __restrict__ out) {
    __shared__ unsigned int sXhi[TILE * LDSW];
    __shared__ unsigned int sXlo[TILE * LDSW];
    __shared__ unsigned int sWhi[TILE * LDSW];
    __shared__ unsigned int sWlo[TILE * LDSW];

    const int t     = threadIdx.x;
    const int lane  = t & 31;
    const int wave  = t >> 5;
    const int l16   = lane & 15;
    const int half  = lane >> 4;
    const int waveM = wave & 3;
    const int waveN = wave >> 2;
    const int rowBase = blockIdx.y * TILE;
    const int colBase = blockIdx.x * TILE;

    v8f acc[2][4];
    const v8f vzero = {0.f, 0.f, 0.f, 0.f, 0.f, 0.f, 0.f, 0.f};
    #pragma unroll
    for (int mi = 0; mi < 2; ++mi)
        #pragma unroll
        for (int ni = 0; ni < 4; ++ni) acc[mi][ni] = vzero;

    float2 rx[8], rw[8];
    auto loadChunk = [&](int kc) {
        const int kOff = kc * KC;
        #pragma unroll
        for (int i = 0; i < 8; ++i) {
            int p = t + i * 256;
            int row = p >> 4, dw = p & 15;
            rx[i] = *(const float2*)&X[(size_t)(rowBase + row) * MDIM + kOff + dw * 2];
            rw[i] = *(const float2*)&W[(size_t)(colBase + row) * MDIM + kOff + dw * 2];
        }
    };
    auto stageChunk = [&]() {
        #pragma unroll
        for (int i = 0; i < 8; ++i) {
            int p = t + i * 256;
            int row = p >> 4, dw = p & 15;
            int idx = row * LDSW + dw;
            unsigned int h, l;
            split2(rx[i].x, rx[i].y, h, l);
            sXhi[idx] = h; sXlo[idx] = l;
            split2(rw[i].x, rw[i].y, h, l);
            sWhi[idx] = h; sWlo[idx] = l;
        }
    };
    auto frag = [&](const unsigned int* base, int row) -> v16bf {
        const unsigned int* rp = base + row * LDSW;
        v8u u;
        #pragma unroll
        for (int v = 0; v < 8; ++v)
            u[v] = rp[(v & 3) + half * 4 + (v >> 2) * 8];
        return __builtin_bit_cast(v16bf, u);
    };

    const int NCH = MDIM / KC;
    loadChunk(0);
    for (int kc = 0; kc < NCH; ++kc) {
        __syncthreads();
        stageChunk();
        __syncthreads();
        if (kc + 1 < NCH) loadChunk(kc + 1);

        v16bf Ah[2], Al[2], Bh[4], Bl[4];
        #pragma unroll
        for (int mi = 0; mi < 2; ++mi) {
            int r = waveM * 32 + mi * 16 + l16;
            Ah[mi] = frag(sXhi, r);
            Al[mi] = frag(sXlo, r);
        }
        #pragma unroll
        for (int ni = 0; ni < 4; ++ni) {
            int r = waveN * 64 + ni * 16 + l16;
            Bh[ni] = frag(sWhi, r);
            Bl[ni] = frag(sWlo, r);
        }
        #pragma unroll
        for (int mi = 0; mi < 2; ++mi) {
            #pragma unroll
            for (int ni = 0; ni < 4; ++ni) {
                acc[mi][ni] = __builtin_amdgcn_wmma_f32_16x16x32_bf16(
                    false, Ah[mi], false, Bh[ni], (short)0, acc[mi][ni], false, false);
                acc[mi][ni] = __builtin_amdgcn_wmma_f32_16x16x32_bf16(
                    false, Ah[mi], false, Bl[ni], (short)0, acc[mi][ni], false, false);
                acc[mi][ni] = __builtin_amdgcn_wmma_f32_16x16x32_bf16(
                    false, Al[mi], false, Bh[ni], (short)0, acc[mi][ni], false, false);
            }
        }
    }

    #pragma unroll
    for (int mi = 0; mi < 2; ++mi) {
        #pragma unroll
        for (int ni = 0; ni < 4; ++ni) {
            const int n = colBase + waveN * 64 + ni * 16 + l16;
            const float w2v = 0.5f * w2[n];
            #pragma unroll
            for (int r = 0; r < 8; ++r) {
                const int m = rowBase + waveM * 32 + mi * 16 + r + 8 * half;
                const float d = acc[mi][ni][r] - 0.5f * x2[m] - w2v;
                out[(size_t)m * NDIM + n] = expf(d);
            }
        }
    }
}

// ------------------------------- launcher ----------------------------------
extern "C" void kernel_launch(void* const* d_in, const int* in_sizes, int n_in,
                              void* d_out, int out_size, void* d_ws, size_t ws_size,
                              hipStream_t stream) {
    const float* X = (const float*)d_in[0];   // [4096, 512]
    const float* W = (const float*)d_in[1];   // [8192, 512]
    float* out = (float*)d_out;               // [4096, 8192]

    const size_t normDW  = BDIM + NDIM;                       // 12288 dwords
    const size_t xPlane  = (size_t)BDIM * (MDIM / 2);         // dwords per X plane
    const size_t wPlane  = (size_t)NDIM * (MDIM / 2);         // dwords per W plane
    const size_t needFast = (normDW + 2 * xPlane + 2 * wPlane) * 4;  // ~24.1 MB

    float* x2 = (float*)d_ws;
    float* w2 = x2 + BDIM;
    dim3 grid(NDIM / TILE, BDIM / TILE);                      // (64, 32)

    if (ws_size >= needFast) {
        unsigned int* Xhi = (unsigned int*)d_ws + normDW;
        unsigned int* Xlo = Xhi + xPlane;
        unsigned int* Whi = Xlo + xPlane;
        unsigned int* Wlo = Whi + wPlane;
        split_norm_kernel<<<(BDIM + NDIM) / 8, 256, 0, stream>>>(
            X, W, x2, w2, Xhi, Xlo, Whi, Wlo);
        rbf_wmma_async_kernel<<<grid, 256, 0, stream>>>(
            Xhi, Xlo, Whi, Wlo, x2, w2, out);
    } else {
        rownorm_kernel<<<(BDIM + NDIM) / 8, 256, 0, stream>>>(X, W, x2, w2);
        rbf_wmma_kernel<<<grid, 256, 0, stream>>>(X, W, x2, w2, out);
    }
}
